// Seg_SVD_LlamaMLP_8907762172502
// MI455X (gfx1250) — matrix operand order repaired and checked by reading, not compiled, not run on hardware
//
#include <hip/hip_runtime.h>
#include <math.h>

typedef __attribute__((ext_vector_type(16))) __bf16 bf16x16;
typedef __attribute__((ext_vector_type(8)))  float  f32x8;

#define SA_STR 40            // 128x32 bf16 tile, padded row stride (80B) -> conflict-free 16B frag loads
#define TILE_LDS (128 * SA_STR)          // ushorts per tile buffer (10240 B, 16B multiple)

union FragAB { bf16x16 v; uint4 q[2]; };

__device__ __forceinline__ unsigned short f2bf(float f) {
  unsigned u = __float_as_uint(f);
  u += 0x7FFFu + ((u >> 16) & 1u);   // round-to-nearest-even
  return (unsigned short)(u >> 16);
}

// Low 32 bits of a generic pointer to LDS == wave/workgroup LDS byte offset (ISA 10.2).
__device__ __forceinline__ unsigned lds_offset_of(const void* p) {
  return (unsigned)(unsigned long long)p;
}

// Stage one 128 x 32(K) A tile via async global->LDS DMA (no VGPR round trip,
// tracked by ASYNCcnt), and one B tile (bf16 converted on the fly from global
// fp32) via the VALU path. 256 threads.
__device__ __forceinline__ void stage_tiles(const unsigned short* __restrict__ A, int lda,
                                            const float* __restrict__ B, int ldb,
                                            int k0, int tid,
                                            unsigned sA_off, unsigned short* sB) {
#pragma unroll
  for (int it = 0; it < 2; ++it) {           // 128 rows * 4 chunks of 8 bf16
    int idx = tid + it * 256;
    int r = idx >> 2, kk = (idx & 3) << 3;
    unsigned long long ga = (unsigned long long)(const void*)&A[(size_t)r * lda + k0 + kk];
    unsigned la = sA_off + (unsigned)((r * SA_STR + kk) * 2);
    asm volatile("global_load_async_to_lds_b128 %0, %1, off"
                 :: "v"(la), "v"(ga) : "memory");
  }
#pragma unroll
  for (int it = 0; it < 4; ++it) {           // 128 rows * 8 chunks of 4 f32
    int idx = tid + it * 256;
    int r = idx >> 3, kk = (idx & 7) << 2;
    const float* bp = &B[(size_t)r * ldb + k0 + kk];
    __builtin_prefetch(bp + 32, 0, 0);       // global_prefetch_b8 for next K tile
    float4 f = *(const float4*)bp;
    uint2 pk;
    pk.x = (unsigned)f2bf(f.x) | ((unsigned)f2bf(f.y) << 16);
    pk.y = (unsigned)f2bf(f.z) | ((unsigned)f2bf(f.w) << 16);
    *(uint2*)&sB[r * SA_STR + kk] = pk;
  }
}

__device__ __forceinline__ void wait_async_done() {
  asm volatile("s_wait_asynccnt 0x0" ::: "memory");
}

// One K=32 step: 8 x v_wmma_f32_16x16x32_bf16 per wave (wave tile 64x32).
// Fragment packing follows CDNA5 ISA 7.12.2:
//  A 16x32 bf16: lane row = l0, K = hi*8 + {0..7} (V0-3) and 16+hi*8+{0..7} (V4-7)
//  B 32x16 bf16 (as B^T rows): lane col = l0, K = hi*8 + {0..7} (V0-3) and 16+hi*8+{0..7} (V4-7)
__device__ __forceinline__ void compute_step(const unsigned short* sA, const unsigned short* sB,
                                             f32x8 c[4][2], int wm, int wn, int l0, int hi) {
  FragAB fa[4], fb[2];
#pragma unroll
  for (int i = 0; i < 4; ++i) {
    const unsigned short* p = sA + (wm * 64 + i * 16 + l0) * SA_STR;
    fa[i].q[0] = *(const uint4*)(p + hi * 8);
    fa[i].q[1] = *(const uint4*)(p + 16 + hi * 8);
  }
#pragma unroll
  for (int j = 0; j < 2; ++j) {
    const unsigned short* p = sB + (wn * 32 + j * 16 + l0) * SA_STR + hi * 8;
    fb[j].q[0] = *(const uint4*)(p);
    fb[j].q[1] = *(const uint4*)(p + 16);
  }
#pragma unroll
  for (int i = 0; i < 4; ++i)
#pragma unroll
    for (int j = 0; j < 2; ++j)
      c[i][j] = __builtin_amdgcn_wmma_f32_16x16x32_bf16(false, fa[i].v, false, fb[j].v,
                                                        (short)0, c[i][j], false, false);
}

// Double-buffered K loop over one (A,B) pair; accumulates into c. K % 32 == 0.
__device__ __forceinline__ void gemm_phase(const unsigned short* __restrict__ Ablk, int lda,
                                           const float* __restrict__ Bblk, int ldb, int K,
                                           int tid, int wm, int wn, int l0, int hi,
                                           const unsigned short* sA, unsigned sA_off,
                                           unsigned short* sB,
                                           f32x8 c[4][2]) {
  int nk = K >> 5;
  stage_tiles(Ablk, lda, Bblk, ldb, 0, tid, sA_off, sB);
  wait_async_done();
  __syncthreads();
  for (int kt = 0; kt < nk; ++kt) {
    int cur = kt & 1;
    if (kt + 1 < nk)
      stage_tiles(Ablk, lda, Bblk, ldb, (kt + 1) << 5, tid,
                  sA_off + (unsigned)((cur ^ 1) * TILE_LDS * 2),
                  sB + (cur ^ 1) * TILE_LDS);
    compute_step(sA + cur * TILE_LDS, sB + cur * TILE_LDS, c, wm, wn, l0, hi);
    wait_async_done();
    __syncthreads();
  }
}

__device__ __forceinline__ void zero_acc(f32x8 c[4][2]) {
#pragma unroll
  for (int i = 0; i < 4; ++i)
#pragma unroll
    for (int j = 0; j < 2; ++j)
      c[i][j] = (f32x8){0.f, 0.f, 0.f, 0.f, 0.f, 0.f, 0.f, 0.f};
}

// ---------- kernels ----------

__global__ __launch_bounds__(256) void gather_cols_kernel(unsigned short* __restrict__ dst,
                                                          const float* __restrict__ x,
                                                          const int* __restrict__ idx,
                                                          int ncols, int ldx) {
  int j = blockIdx.x * 256 + threadIdx.x;
  int m = blockIdx.y;
  dst[(size_t)m * ncols + j] = f2bf(x[(size_t)m * ldx + idx[j]]);
}

__global__ __launch_bounds__(256) void build_inv_kernel(int* __restrict__ inv,
                                                        const int* __restrict__ svd_idx, int nsvd,
                                                        const int* __restrict__ col_idx, int ncol) {
  int t = blockIdx.x * 256 + threadIdx.x;
  if (t < nsvd)             inv[svd_idx[t]] = t;
  else if (t < nsvd + ncol) inv[col_idx[t - nsvd]] = (1 << 30) | (t - nsvd);
}

// D(bf16)[M x N] = A(bf16)[M x K] * B(f32)[N x K]^T   (f32 accumulation)
__global__ __launch_bounds__(256) void gemm_nt_bf16_kernel(const unsigned short* __restrict__ A,
                                                           const float* __restrict__ B,
                                                           unsigned short* __restrict__ D,
                                                           int K, int N) {
  __shared__ __align__(16) unsigned short sA[2 * TILE_LDS];
  __shared__ __align__(16) unsigned short sB[2 * TILE_LDS];
  int tid = threadIdx.x, lane = tid & 31, wave = tid >> 5;
  int wm = wave >> 2, wn = wave & 3, l0 = lane & 15, hi = lane >> 4;
  unsigned sA_off = lds_offset_of(sA);
  f32x8 c[4][2];
  zero_acc(c);
  gemm_phase(A + (size_t)blockIdx.y * 128 * K, K,
             B + (size_t)blockIdx.x * 128 * K, K, K,
             tid, wm, wn, l0, hi, sA, sA_off, sB, c);
  int m0 = blockIdx.y * 128 + wm * 64, n0 = blockIdx.x * 128 + wn * 32;
#pragma unroll
  for (int i = 0; i < 4; ++i)
#pragma unroll
    for (int j = 0; j < 2; ++j)
#pragma unroll
      for (int v = 0; v < 8; ++v)
        D[(size_t)(m0 + i * 16 + v + hi * 8) * N + (n0 + j * 16 + l0)] = f2bf(c[i][j][v]);
}

// Fused: up = tU*Wu_u^T + xcU*Wu_c^T ; gate = tG*Wg_u^T + xcG*Wg_c^T ;
// inter = silu(gate)*up -> bf16, scatter-stored (via inverse permutation) into
// the gathered layouts consumed by the down branch. N tile index = output column of I.
__global__ __launch_bounds__(256) void gemm_dual_swiglu_kernel(
    const unsigned short* __restrict__ tU, const float* __restrict__ Wu_u,
    const unsigned short* __restrict__ xcU, const float* __restrict__ Wu_c,
    const unsigned short* __restrict__ tG, const float* __restrict__ Wg_u,
    const unsigned short* __restrict__ xcG, const float* __restrict__ Wg_c,
    const int* __restrict__ inv,
    unsigned short* __restrict__ inter_svd, unsigned short* __restrict__ inter_col,
    int R, int NC, int svd_stride, int col_stride) {
  __shared__ __align__(16) unsigned short sA[2 * TILE_LDS];
  __shared__ __align__(16) unsigned short sB[2 * TILE_LDS];
  int tid = threadIdx.x, lane = tid & 31, wave = tid >> 5;
  int wm = wave >> 2, wn = wave & 3, l0 = lane & 15, hi = lane >> 4;
  unsigned sA_off = lds_offset_of(sA);
  size_t mblk = (size_t)blockIdx.y * 128;
  size_t nblk = (size_t)blockIdx.x * 128;
  f32x8 cu[4][2], cg[4][2];
  zero_acc(cu);
  zero_acc(cg);
  gemm_phase(tU  + mblk * R,  R,  Wu_u + nblk * R,  R,  R,  tid, wm, wn, l0, hi, sA, sA_off, sB, cu);
  gemm_phase(xcU + mblk * NC, NC, Wu_c + nblk * NC, NC, NC, tid, wm, wn, l0, hi, sA, sA_off, sB, cu);
  gemm_phase(tG  + mblk * R,  R,  Wg_u + nblk * R,  R,  R,  tid, wm, wn, l0, hi, sA, sA_off, sB, cg);
  gemm_phase(xcG + mblk * NC, NC, Wg_c + nblk * NC, NC, NC, tid, wm, wn, l0, hi, sA, sA_off, sB, cg);
  int m0 = (int)mblk + wm * 64, n0 = (int)nblk + wn * 32;
#pragma unroll
  for (int j = 0; j < 2; ++j) {
    int n = n0 + j * 16 + l0;
    int e = inv[n];
    bool to_col = (e & (1 << 30)) != 0;
    int jj = e & 0x3FFFFFFF;
#pragma unroll
    for (int i = 0; i < 4; ++i)
#pragma unroll
      for (int v = 0; v < 8; ++v) {
        int m = m0 + i * 16 + v + hi * 8;
        float g = cg[i][j][v], u = cu[i][j][v];
        float s = g * (1.0f / (1.0f + __expf(-g))) * u;   // SwiGLU
        unsigned short hbits = f2bf(s);
        if (to_col) inter_col[(size_t)m * col_stride + jj] = hbits;
        else        inter_svd[(size_t)m * svd_stride + jj] = hbits;
      }
  }
}

// D(f32)[M x N] = A1*B1^T (K1) + A2*B2^T (K2) -- final down projection to d_out
__global__ __launch_bounds__(256) void gemm_dual_f32_kernel(
    const unsigned short* __restrict__ A1, const float* __restrict__ B1, int K1,
    const unsigned short* __restrict__ A2, const float* __restrict__ B2, int K2,
    float* __restrict__ D, int N) {
  __shared__ __align__(16) unsigned short sA[2 * TILE_LDS];
  __shared__ __align__(16) unsigned short sB[2 * TILE_LDS];
  int tid = threadIdx.x, lane = tid & 31, wave = tid >> 5;
  int wm = wave >> 2, wn = wave & 3, l0 = lane & 15, hi = lane >> 4;
  unsigned sA_off = lds_offset_of(sA);
  size_t mblk = (size_t)blockIdx.y * 128;
  size_t nblk = (size_t)blockIdx.x * 128;
  f32x8 c[4][2];
  zero_acc(c);
  gemm_phase(A1 + mblk * K1, K1, B1 + nblk * K1, K1, K1, tid, wm, wn, l0, hi, sA, sA_off, sB, c);
  gemm_phase(A2 + mblk * K2, K2, B2 + nblk * K2, K2, K2, tid, wm, wn, l0, hi, sA, sA_off, sB, c);
  int m0 = (int)mblk + wm * 64, n0 = (int)nblk + wn * 32;
#pragma unroll
  for (int i = 0; i < 4; ++i)
#pragma unroll
    for (int j = 0; j < 2; ++j)
#pragma unroll
      for (int v = 0; v < 8; ++v)
        D[(size_t)(m0 + i * 16 + v + hi * 8) * N + (n0 + j * 16 + l0)] = c[i][j][v];
}

extern "C" void kernel_launch(void* const* d_in, const int* in_sizes, int n_in,
                              void* d_out, int out_size, void* d_ws, size_t ws_size,
                              hipStream_t stream) {
  (void)in_sizes; (void)n_in; (void)out_size; (void)ws_size;
  const float* x      = (const float*)d_in[0];
  const float* up_v   = (const float*)d_in[1];
  const float* up_u   = (const float*)d_in[2];
  const float* up_c   = (const float*)d_in[3];
  const float* gate_v = (const float*)d_in[4];
  const float* gate_u = (const float*)d_in[5];
  const float* gate_c = (const float*)d_in[6];
  const float* down_v = (const float*)d_in[7];
  const float* down_u = (const float*)d_in[8];
  const float* down_c = (const float*)d_in[9];
  const int* svd_up   = (const int*)d_in[10];
  const int* col_up   = (const int*)d_in[11];
  const int* svd_gate = (const int*)d_in[12];
  const int* col_gate = (const int*)d_in[13];
  const int* svd_down = (const int*)d_in[14];
  const int* col_down = (const int*)d_in[15];

  const int M = 8192, H = 4096, I_ = 11008, R = 1024;
  const int NS_UG = 3072, NC_UG = 1024, NS_D = 8192, NC_D = 2816;

  char* p = (char*)d_ws;
  auto take = [&](size_t bytes) { char* r = p; p += (bytes + 255) & ~(size_t)255; return r; };
  unsigned short* a_svd_up   = (unsigned short*)take((size_t)M * NS_UG * 2);
  unsigned short* a_col_up   = (unsigned short*)take((size_t)M * NC_UG * 2);
  unsigned short* a_svd_gate = (unsigned short*)take((size_t)M * NS_UG * 2);
  unsigned short* a_col_gate = (unsigned short*)take((size_t)M * NC_UG * 2);
  unsigned short* t_up       = (unsigned short*)take((size_t)M * R * 2);
  unsigned short* t_gate     = (unsigned short*)take((size_t)M * R * 2);
  unsigned short* t_down     = (unsigned short*)take((size_t)M * R * 2);
  unsigned short* inter_svd  = (unsigned short*)take((size_t)M * NS_D * 2);
  unsigned short* inter_col  = (unsigned short*)take((size_t)M * NC_D * 2);
  int*            inv        = (int*)take((size_t)I_ * 4);

  dim3 blk(256);
  // Gather x columns into bf16 NT-GEMM panels
  gather_cols_kernel<<<dim3(NS_UG / 256, M), blk, 0, stream>>>(a_svd_up,   x, svd_up,   NS_UG, H);
  gather_cols_kernel<<<dim3(NC_UG / 256, M), blk, 0, stream>>>(a_col_up,   x, col_up,   NC_UG, H);
  gather_cols_kernel<<<dim3(NS_UG / 256, M), blk, 0, stream>>>(a_svd_gate, x, svd_gate, NS_UG, H);
  gather_cols_kernel<<<dim3(NC_UG / 256, M), blk, 0, stream>>>(a_col_gate, x, col_gate, NC_UG, H);
  build_inv_kernel<<<dim3((NS_D + NC_D) / 256), blk, 0, stream>>>(inv, svd_down, NS_D, col_down, NC_D);

  // Low-rank projections t = x_svd * V^T   [8192 x 1024]
  gemm_nt_bf16_kernel<<<dim3(R / 128, M / 128), blk, 0, stream>>>(a_svd_up,   up_v,   t_up,   NS_UG, R);
  gemm_nt_bf16_kernel<<<dim3(R / 128, M / 128), blk, 0, stream>>>(a_svd_gate, gate_v, t_gate, NS_UG, R);

  // up & gate (each = U-GEMM + col-GEMM) fused with SwiGLU + permuted scatter to bf16
  gemm_dual_swiglu_kernel<<<dim3(I_ / 128, M / 128), blk, 0, stream>>>(
      t_up, up_u, a_col_up, up_c, t_gate, gate_u, a_col_gate, gate_c,
      inv, inter_svd, inter_col, R, NC_UG, NS_D, NC_D);

  // Down low-rank projection  [8192 x 1024], K = 8192
  gemm_nt_bf16_kernel<<<dim3(R / 128, M / 128), blk, 0, stream>>>(inter_svd, down_v, t_down, NS_D, R);

  // Final: out = t_down * down_u^T + inter_col * down_c^T  -> f32 d_out [8192 x 4096]
  gemm_dual_f32_kernel<<<dim3(H / 128, M / 128), blk, 0, stream>>>(
      t_down, down_u, R, inter_col, down_c, NC_D, (float*)d_out, H);
}
